// SpatialCL_59614146068755
// MI455X (gfx1250) — compile-verified, matrix operand order checked by reading, not run on hardware
//
#include <hip/hip_runtime.h>
#include <hip/hip_bf16.h>

typedef _Float16     v16h __attribute__((ext_vector_type(16)));
typedef _Float16     v8h  __attribute__((ext_vector_type(8)));
typedef float        v8f  __attribute__((ext_vector_type(8)));
typedef unsigned int v4u  __attribute__((ext_vector_type(4)));
typedef int          v8i  __attribute__((ext_vector_type(8)));
typedef int          v4i  __attribute__((ext_vector_type(4)));

#define BQ     512
#define DDIM   256
#define NTOT   131072

#define MSLAB    2048              /* memory rows per block            */
#define MSTAGE   128               /* rows per TDM stage               */
#define NSTAGES  (MSLAB / MSTAGE)  /* 16                               */
#define ROWPITCH 544               /* 512B row + 32B TDM pad (2-way bc)*/
#define STAGE_BYTES (MSTAGE * ROWPITCH)      /* 69632  */
#define CA_OFF      (2 * STAGE_BYTES)        /* 139264 */
#define SMEM_BYTES  (CA_OFF + 2 * MSTAGE * 4)/* 140288 < 160KB -> 2 blk/WGP */

/* ws layout */
#define QF16_OFF   0
#define QF16_BYTES (BQ * DDIM * 2)                 /* 262144  */
#define QATTR_OFF  (QF16_OFF + QF16_BYTES)         /* +2048   */
#define CATTR_OFF  (QATTR_OFF + 2048)              /* +524288 */
#define MF16_OFF   (CATTR_OFF + NTOT * 4)          /* +64MB   */

#define QSCALE 7.2134752044448170f   /* 5 * log2(e): folds temp + exp->exp2 */

#if defined(__gfx1250__) && __has_builtin(__builtin_amdgcn_exp2f)
#define FAST_EXP2(x) __builtin_amdgcn_exp2f(x)   /* raw v_exp_f32 */
#else
#define FAST_EXP2(x) exp2f(x)
#endif

// ---------------------------------------------------------------------------
// Prep 1: normalize q rows, fold softmax scale, -> f16; pack attrs; zero out.
// ---------------------------------------------------------------------------
__global__ __launch_bounds__(256) void prep_q_kernel(
    const float* __restrict__ q, const int* __restrict__ bag,
    const int* __restrict__ xc, const int* __restrict__ yc,
    _Float16* __restrict__ qf16, int* __restrict__ qattr,
    float* __restrict__ out)
{
    __shared__ float red[256];
    const int b = blockIdx.x;
    const int t = threadIdx.x;
    const float v = q[b * DDIM + t];
    red[t] = v * v;
    __syncthreads();
    #pragma unroll
    for (int s = 128; s > 0; s >>= 1) {
        if (t < s) red[t] += red[t + s];
        __syncthreads();
    }
    const float rn = rsqrtf(red[0]) * QSCALE;
    qf16[b * DDIM + t] = (_Float16)(v * rn);
    if (t == 0) {
        qattr[b] = (bag[b] << 20) | (xc[b] << 10) | yc[b];
        out[b]   = 0.0f;   // harness poisons d_out; re-zero every call
    }
}

// ---------------------------------------------------------------------------
// Prep 2: pack memory-bank attrs (bag,x,y) into one int per entry.
// ---------------------------------------------------------------------------
__global__ __launch_bounds__(256) void prep_c_kernel(
    const int* __restrict__ bag, const int* __restrict__ xc,
    const int* __restrict__ yc, int* __restrict__ cattr, int n)
{
    const int i = blockIdx.x * 256 + threadIdx.x;
    if (i < n) cattr[i] = (bag[i] << 20) | (xc[i] << 10) | yc[i];
}

// ---------------------------------------------------------------------------
// Prep 3: memory bank f32 -> f16 (one shot; enables TDM streaming of A tiles).
// ---------------------------------------------------------------------------
__global__ __launch_bounds__(256) void prep_m_kernel(
    const float* __restrict__ mf, _Float16* __restrict__ mh)
{
    const size_t i = ((size_t)blockIdx.x * 256 + threadIdx.x) * 8;
    const float4* p = (const float4*)(mf + i);
    const float4 a = p[0], b = p[1];
    v8h h;
    h[0] = (_Float16)a.x; h[1] = (_Float16)a.y;
    h[2] = (_Float16)a.z; h[3] = (_Float16)a.w;
    h[4] = (_Float16)b.x; h[5] = (_Float16)b.y;
    h[6] = (_Float16)b.z; h[7] = (_Float16)b.w;
    *(v8h*)(mh + i) = h;
}

// ---------------------------------------------------------------------------
// TDM: DMA a [128 x 256] f16 tile (row-major) into LDS with 32B/row padding.
// ---------------------------------------------------------------------------
__device__ __forceinline__ void tdm_stage_load(const _Float16* gsrc,
                                               unsigned ldsoff)
{
    const unsigned long long ga = (unsigned long long)(uintptr_t)gsrc;
    v4u g0;
    g0[0] = 1u;                                       // count=1 (user desc)
    g0[1] = ldsoff;                                   // lds_addr (bytes)
    g0[2] = (unsigned)(ga & 0xFFFFFFFFu);             // global_addr lo
    g0[3] = (unsigned)((ga >> 32) & 0x1FFFFFFu)       // global_addr hi (57b)
          | (2u << 30);                               // type=2 (image)
    v8i g1;
    g1[0] = (int)((1u << 16)        // data_size = 2B
                | (1u << 20)        // pad_enable
                | (6u << 22)        // pad_interval: 2^(6+1)=128 DW (=512B row)
                | (7u << 25));      // pad_amount: 8 DW = 32B
    g1[1] = (int)(256u << 16);      // tensor_dim0 = 256 (bits 48..63)
    g1[2] = (int)(0);               // dim0 hi | tensor_dim1[15:0] (131072->0)
    g1[3] = (int)(2u | (256u << 16));  // tensor_dim1[31:16]=2 | tile_dim0=256
    g1[4] = (int)MSTAGE;            // tile_dim1 = 128, tile_dim2 = 0
    g1[5] = 256;                    // tensor_dim0_stride = 256 elements
    g1[6] = 0;
    g1[7] = 0;
    v4i g2 = {0, 0, 0, 0};
    v4i g3 = {0, 0, 0, 0};
    v8i g4 = {0, 0, 0, 0, 0, 0, 0, 0};
    __builtin_amdgcn_tensor_load_to_lds(g0, g1, g2, g3, g4, 0);
}

// ---------------------------------------------------------------------------
// Main: transposed WMMA GEMM. M = memory rows (TDM-streamed), N = queries
// (register-resident B frags). Row-sum collapses to a per-lane register
// accumulator. A-frags preloaded into distinct regs so LDS loads pipeline
// ahead of the WMMA chain instead of serializing on dscnt==0.
// ---------------------------------------------------------------------------
__global__ __launch_bounds__(256) void spatialcl_main_kernel(
    const _Float16* __restrict__ mf16, const _Float16* __restrict__ qf16,
    const int* __restrict__ qattr, const int* __restrict__ cattr,
    float* __restrict__ out)
{
    extern __shared__ char smem[];
    int* cal = (int*)(smem + CA_OFF);          // [2][128] staged mem attrs

    const int tid  = threadIdx.x;
    const int wave = tid >> 5;
    const int lane = tid & 31;
    const int l16  = lane & 15;
    const int hi   = lane >> 4;                // 0: lanes 0-15, 1: 16-31

    const int slab   = blockIdx.x >> 2;
    const int qchunk = blockIdx.x & 3;
    const int slabBase = slab * MSLAB;
    const int qbase    = (qchunk * 8 + wave) * 16;

    // ---- Persistent per-wave B fragments: 16 queries x K=256 (64 VGPRs) ---
    const _Float16* qrow = qf16 + (size_t)(qbase + l16) * DDIM + hi * 16;
    v16h bfrag[8];
    #pragma unroll
    for (int ks = 0; ks < 8; ++ks)
        bfrag[ks] = *(const v16h*)(qrow + ks * 32);

    // ---- Per-lane query attrs (fixed for whole kernel) --------------------
    const int qa = qattr[qbase + l16];
    const int qx = (qa >> 10) & 1023;
    const int qy = qa & 1023;

    float acc = 0.0f;

    // ---- Prime the TDM pipeline ------------------------------------------
    if (tid < 32)
        tdm_stage_load(mf16 + (size_t)slabBase * DDIM, (unsigned)(uintptr_t)smem);

    for (int s = 0; s < NSTAGES; ++s) {
        if (tid < 32) {
            if (s + 1 < NSTAGES) {
                tdm_stage_load(mf16 + (size_t)(slabBase + (s + 1) * MSTAGE) * DDIM,
                               (unsigned)(uintptr_t)smem + ((s + 1) & 1) * STAGE_BYTES);
                __builtin_amdgcn_s_wait_tensorcnt(1);   // stage s complete
            } else {
                __builtin_amdgcn_s_wait_tensorcnt(0);
            }
        }
        if (tid < MSTAGE)
            cal[(s & 1) * MSTAGE + tid] = cattr[slabBase + s * MSTAGE + tid];
        __syncthreads();

        const char* buf    = smem + (s & 1) * STAGE_BYTES;
        const int*  calbuf = cal + (s & 1) * MSTAGE;

        #pragma unroll 1
        for (int mt = 0; mt < 8; ++mt) {            // 8 tiles of 16 mem rows
            // Preload all 8 A-frags (distinct regs -> DS loads pipeline)
            const char* arow = buf + (mt * 16 + l16) * ROWPITCH + hi * 32;
            v16h afrag[8];
            #pragma unroll
            for (int ks = 0; ks < 8; ++ks)
                afrag[ks] = *(const v16h*)(arow + ks * 64);
            // Mem-row attrs for this lane's 8 rows: one 32B LDS load
            const v8i calist = *(const v8i*)(calbuf + mt * 16 + hi * 8);

            v8f c = {0.f, 0.f, 0.f, 0.f, 0.f, 0.f, 0.f, 0.f};
            #pragma unroll
            for (int ks = 0; ks < 8; ++ks)
                c = __builtin_amdgcn_wmma_f32_16x16x32_f16(
                        false, afrag[ks], false, bfrag[ks], (short)0, c,
                        false, false);

            // Fused epilogue: mask + raw v_exp_f32 + per-lane accumulate
            #pragma unroll
            for (int r = 0; r < 8; ++r) {
                const int ca = calist[r];
                const int dx = ((ca >> 10) & 1023) - qx;
                const int dy = (ca & 1023) - qy;
                const int d2 = dx * dx + dy * dy;
                const bool ok = ((unsigned)(qa ^ ca) < (1u << 20)) &&
                                ((unsigned)(d2 - 1) < 3u);
                // masked -> 2^-128 == 0.0f in round-trip; |err| < 4e-34 total
                const float t = ok ? c[r] : -128.0f;
                acc += FAST_EXP2(t);
            }
        }
        __syncthreads();
    }

    // ---- Final reduction: combine the two lane halves, one atomic/query ---
    acc += __shfl_xor(acc, 16, 32);
    if (lane < 16)
        atomicAdd(&out[qbase + lane], acc);
}

// ---------------------------------------------------------------------------
extern "C" void kernel_launch(void* const* d_in, const int* in_sizes, int n_in,
                              void* d_out, int out_size, void* d_ws, size_t ws_size,
                              hipStream_t stream) {
    const float* q        = (const float*)d_in[0];
    const float* memory   = (const float*)d_in[1];
    const int*   bag_idx  = (const int*)d_in[2];
    const int*   x_coord  = (const int*)d_in[3];
    const int*   y_coord  = (const int*)d_in[4];
    const int*   bag_idxs = (const int*)d_in[5];
    const int*   x_coords = (const int*)d_in[6];
    const int*   y_coords = (const int*)d_in[7];
    float* out = (float*)d_out;

    char* ws = (char*)d_ws;
    _Float16* qf16  = (_Float16*)(ws + QF16_OFF);
    int*      qattr = (int*)(ws + QATTR_OFF);
    int*      cattr = (int*)(ws + CATTR_OFF);
    _Float16* mf16  = (_Float16*)(ws + MF16_OFF);

    prep_q_kernel<<<BQ, 256, 0, stream>>>(q, bag_idx, x_coord, y_coord,
                                          qf16, qattr, out);
    prep_c_kernel<<<NTOT / 256, 256, 0, stream>>>(bag_idxs, x_coords, y_coords,
                                                  cattr, NTOT);
    prep_m_kernel<<<NTOT * DDIM / 8 / 256, 256, 0, stream>>>(memory, mf16);
    spatialcl_main_kernel<<<(NTOT / MSLAB) * 4, 256, SMEM_BYTES, stream>>>(
        mf16, qf16, qattr, cattr, out);
}
